// CachedGemmaAttention_5866925326867
// MI455X (gfx1250) — matrix operand order repaired and checked by reading, not compiled, not run on hardware
//
#include <hip/hip_runtime.h>
#include <stdint.h>

typedef __attribute__((ext_vector_type(16))) _Float16 v16h;
typedef __attribute__((ext_vector_type(8)))  float    v8f;
typedef __attribute__((ext_vector_type(2)))  _Float16 h2;

#define B_   2
#define T_   2048
#define D_   2304
#define H_   256
#define NQ_  8
#define NKV_ 4
#define WIN_ 1024
#define CAP_ 50.0f

// D = A(16x32 f16) * B(32x16 f16) + C(16x16 f32)
__device__ __forceinline__ v8f wmma16(v16h a, v16h b, v8f c) {
  return __builtin_amdgcn_wmma_f32_16x16x32_f16(false, a, false, b, (short)0, c,
                                                false, false);
}

union FragA { v16h h; uint32_t u[8]; };   // A frag: K-pairs contiguous per VGPR
union FragB { v16h h; uint4 q[2]; };      // B frag: 16 contiguous K values/lane

// A-frag (16x32 f16) lane layout: M = lane%16; VGPR r holds K = kb, kb+1
__device__ __forceinline__ int a_kbase(int r, int lg) {
  return 2 * r + 16 * lg;
}

__device__ __forceinline__ void load_fragB(FragB& f, const _Float16* p) {
  f.q[0] = *(const uint4*)p;
  f.q[1] = *(const uint4*)(p + 8);
}

// branch-free soft-cap: CAP * tanh(x/CAP) = CAP * (1 - 2/(e^{2x/CAP}+1))
__device__ __forceinline__ float softcap(float x) {
  float e = __expf(x * (2.0f / CAP_));
  return CAP_ * (1.0f - 2.0f * __builtin_amdgcn_rcpf(e + 1.0f));
}

// ---------------------------------------------------------------------------
// Kernel 0a: x f32 -> f16, coalesced (4 elements/thread)
// ---------------------------------------------------------------------------
__global__ __launch_bounds__(256)
void xconv_kernel(const float* __restrict__ src, _Float16* __restrict__ dst) {
  const size_t i = ((size_t)blockIdx.x * 256 + threadIdx.x) * 4;
  float4 v = *(const float4*)(src + i);
  h2 a, b;
  a[0] = (_Float16)v.x; a[1] = (_Float16)v.y;
  b[0] = (_Float16)v.z; b[1] = (_Float16)v.w;
  *(h2*)(dst + i)     = a;
  *(h2*)(dst + i + 2) = b;
}

// ---------------------------------------------------------------------------
// Kernel 0b: one-shot weight transpose + f32->f16 convert via LDS 32x33 tile.
// src: [head][R][C] f32  ->  dst: [head][C][R] f16.  R,C multiples of 32.
// block = 256 (32x8), grid = (C/32, R/32, heads)
// ---------------------------------------------------------------------------
__global__ __launch_bounds__(256)
void wtrans_kernel(const float* __restrict__ src, _Float16* __restrict__ dst,
                   int R, int C) {
  __shared__ _Float16 tile[32][33];
  const int tx = threadIdx.x & 31, ty = threadIdx.x >> 5;
  const int c0 = blockIdx.x * 32;
  const int r0 = blockIdx.y * 32;
  const float* s = src + (size_t)blockIdx.z * R * C;
  _Float16*    d = dst + (size_t)blockIdx.z * R * C;
#pragma unroll
  for (int k = 0; k < 4; k++)
    tile[ty + 8 * k][tx] = (_Float16)s[(size_t)(r0 + ty + 8 * k) * C + c0 + tx];
  __syncthreads();
#pragma unroll
  for (int k = 0; k < 4; k++)
    d[(size_t)(c0 + ty + 8 * k) * R + r0 + tx] = tile[tx][ty + 8 * k];
}

// ---------------------------------------------------------------------------
// Kernel 1: fused QKV projection (WMMA) + RoPE (+ q-scale). V stored as V^T.
// x pre-converted f16; weights pre-transposed f16: wt[proj_head][H][D].
// grid = B * 16 projheads * T/16, block = 1 wave (32)
// ---------------------------------------------------------------------------
__global__ __launch_bounds__(32)
void qkv_rope_kernel(const _Float16* __restrict__ xh,  // [B,T,D]
                     const _Float16* __restrict__ wt,  // [16][H][D]
                     _Float16* __restrict__ q_h,       // [B,T,NQ,H]
                     _Float16* __restrict__ k_h,       // [B,T,NKV,H]
                     _Float16* __restrict__ v_t) {     // [B,NKV,H,T]
  const int lane = threadIdx.x;
  const int l16 = lane & 15, lg = lane >> 4;
  const int tb = blockIdx.x & 127;
  const int ph = (blockIdx.x >> 7) & 15;
  const int bb = blockIdx.x >> 11;
  const int t0 = tb * 16;

  int head, kind;                       // 0=q, 1=k, 2=v
  if (ph < NQ_)             { kind = 0; head = ph; }
  else if (ph < NQ_ + NKV_) { kind = 1; head = ph - NQ_; }
  else                      { kind = 2; head = ph - NQ_ - NKV_; }
  const _Float16* W = wt + (size_t)ph * H_ * D_;   // transposed: [H][D]

  v8f zero = {};
  v8f acc[16];
#pragma unroll
  for (int i = 0; i < 16; i++) acc[i] = zero;

  const _Float16* xrow = xh + ((size_t)bb * T_ + (t0 + l16)) * D_;

  for (int kk0 = 0; kk0 < D_; kk0 += 32) {
    FragA fa;                                      // x tile: 8x b32 pair loads
#pragma unroll
    for (int r = 0; r < 8; r++)
      fa.u[r] = *(const uint32_t*)(xrow + kk0 + a_kbase(r, lg));
    const _Float16* wbase = W + (size_t)l16 * D_ + kk0 + 16 * lg;
    __builtin_prefetch(wbase + 32, 0, 1);
#pragma unroll
    for (int nt = 0; nt < 16; nt++) {
      FragB fb;                                    // 2x global_load_b128
      load_fragB(fb, wbase + (size_t)nt * 16 * D_);
      acc[nt] = wmma16(fa.h, fb.h, acc[nt]);
    }
  }

  if (kind == 2) {                                 // V: store transposed
#pragma unroll
    for (int nt = 0; nt < 16; nt++) {
      int h = nt * 16 + l16;
      _Float16* dst =
          v_t + (((size_t)bb * NKV_ + head) * H_ + h) * T_ + t0 + 8 * lg;
#pragma unroll
      for (int r = 0; r < 8; r++) dst[r] = (_Float16)acc[nt][r];
    }
  } else {                                         // Q/K: RoPE (+ scale)
    const float qscale = (kind == 0) ? 0.0625f : 1.0f;  // 1/sqrt(256)
    const int rowstride = (kind == 0 ? NQ_ : NKV_) * H_;
    _Float16* outp = (kind == 0)
        ? q_h + (((size_t)bb * T_ + t0) * NQ_ + head) * H_
        : k_h + (((size_t)bb * T_ + t0) * NKV_ + head) * H_;
#pragma unroll
    for (int jt = 0; jt < 8; jt++) {
#pragma unroll
      for (int r = 0; r < 8; r++) {
        float a  = acc[jt][r];
        float b2 = acc[jt + 8][r];
        int   i  = jt * 16 + l16;                  // freq index 0..127
        // inv_freq = 10000^(-i/128) = exp(-i * ln(1e4)/128)
        float invf = __expf(-(float)i * 0.0719558029f);
        float ang  = (float)(t0 + r + 8 * lg) * invf;
        float sn, cs;
        __sincosf(ang, &sn, &cs);
        h2 hv;
        hv[0] = (_Float16)((a * cs - b2 * sn) * qscale);   // col 2i
        hv[1] = (_Float16)((b2 * cs + a * sn) * qscale);   // col 2i+1
        *(h2*)(outp + (size_t)(r + 8 * lg) * rowstride + 2 * i) = hv;
      }
    }
  }
}

// ---------------------------------------------------------------------------
// Kernel 2: flash attention, sliding-window causal, tanh soft-cap, GQA.
// grid = B * NQ * T/16, block = 1 wave (32)
// ---------------------------------------------------------------------------
__global__ __launch_bounds__(32)
void attn_kernel(const _Float16* __restrict__ q_h,
                 const _Float16* __restrict__ k_h,
                 const _Float16* __restrict__ v_t,
                 _Float16* __restrict__ ao) {       // [B,T,NQ,H]
  __shared__ _Float16 pbuf[16 * 32];                // P tile relayout
  const int lane = threadIdx.x, l16 = lane & 15, lg = lane >> 4;
  const int tb = blockIdx.x & 127;
  const int nh = (blockIdx.x >> 7) & 7;
  const int bb = blockIdx.x >> 10;
  const int kh = nh >> 1;                           // group = NQ/NKV = 2
  const int t0 = tb * 16;

  v8f zero = {};
  v8f acc[16];
#pragma unroll
  for (int i = 0; i < 16; i++) acc[i] = zero;
  float mrow[8], lrow[8];
#pragma unroll
  for (int r = 0; r < 8; r++) { mrow[r] = -1e30f; lrow[r] = 0.f; }

  const _Float16* qbase =
      q_h + (((size_t)bb * T_ + (t0 + l16)) * NQ_ + nh) * H_;

  int lo = t0 - (WIN_ - 1);
  if (lo < 0) lo = 0;
  const int sb0 = lo & ~31;

  for (int sb = sb0; sb <= t0 + 15; sb += 32) {
    // ---- logits: Q(16x256) * K^T(256x32) -> two 16x16 tiles
    v8f c0 = zero, c1 = zero;
    const _Float16* kbase =
        k_h + (((size_t)bb * T_ + (sb + l16)) * NKV_ + kh) * H_;
    __builtin_prefetch(kbase + (size_t)32 * NKV_ * H_, 0, 1);
#pragma unroll
    for (int ks = 0; ks < 8; ks++) {
      const int kk0 = ks * 32;
      FragA fq;
#pragma unroll
      for (int r = 0; r < 8; r++)
        fq.u[r] = *(const uint32_t*)(qbase + kk0 + a_kbase(r, lg));
      FragB fk0, fk1;                               // K^T frags (H contiguous)
      const _Float16* kp0 = kbase + kk0 + 16 * lg;
      load_fragB(fk0, kp0);
      load_fragB(fk1, kp0 + (size_t)16 * NKV_ * H_);
      c0 = wmma16(fq.h, fk0.h, c0);
      c1 = wmma16(fq.h, fk1.h, c1);
    }
    // ---- soft-cap, mask, online softmax
#pragma unroll
    for (int r = 0; r < 8; r++) {
      const int irow = t0 + r + 8 * lg;
      const int j0 = sb + l16, j1 = j0 + 16;
      float x0 = softcap(c0[r]);
      float x1 = softcap(c1[r]);
      const bool v0 = (j0 <= irow) && (irow - j0 < WIN_);
      const bool v1 = (j1 <= irow) && (irow - j1 < WIN_);
      float bm = fmaxf(v0 ? x0 : -1e30f, v1 ? x1 : -1e30f);
#pragma unroll
      for (int off = 1; off < 16; off <<= 1)
        bm = fmaxf(bm, __shfl_xor(bm, off, 32));
      const float mnew  = fmaxf(mrow[r], bm);
      const float p0    = v0 ? __expf(x0 - mnew) : 0.f;
      const float p1    = v1 ? __expf(x1 - mnew) : 0.f;
      const float alpha = __expf(mrow[r] - mnew);
      float ps = p0 + p1;
#pragma unroll
      for (int off = 1; off < 16; off <<= 1)
        ps += __shfl_xor(ps, off, 32);
      lrow[r] = lrow[r] * alpha + ps;
      mrow[r] = mnew;
#pragma unroll
      for (int ht = 0; ht < 16; ht++) acc[ht][r] *= alpha;
      pbuf[(r + 8 * lg) * 32 + l16]      = (_Float16)p0;
      pbuf[(r + 8 * lg) * 32 + 16 + l16] = (_Float16)p1;
    }
    // ---- P(16x32) * V(32x256): A frag from LDS, B frags from V^T
    FragA fp;
#pragma unroll
    for (int r = 0; r < 8; r++)
      fp.u[r] = *(const uint32_t*)&pbuf[l16 * 32 + a_kbase(r, lg)];
    const _Float16* vbase =
        v_t + ((size_t)bb * NKV_ + kh) * H_ * T_ + sb + 16 * lg;
#pragma unroll
    for (int ht = 0; ht < 16; ht++) {
      FragB fv;
      load_fragB(fv, vbase + (size_t)(ht * 16 + l16) * T_);
      acc[ht] = wmma16(fp.h, fv.h, acc[ht]);
    }
  }
  // ---- normalize and store (causal: every row attends itself -> l > 0)
#pragma unroll
  for (int r = 0; r < 8; r++) {
    const float inv = __builtin_amdgcn_rcpf(lrow[r]);
    const int trow = t0 + r + 8 * lg;
    _Float16* op = ao + (((size_t)bb * T_ + trow) * NQ_ + nh) * H_ + l16;
#pragma unroll
    for (int ht = 0; ht < 16; ht++)
      op[ht * 16] = (_Float16)(acc[ht][r] * inv);
  }
}

// ---------------------------------------------------------------------------
// Kernel 3: output projection  [B*T,2048] x [2048,2304] -> f32 out
// Weights pre-transposed f16: wot[D][2048].  16x128 tile per wave.
// grid = (B*T/16) * (D/128), block = 1 wave (32)
// ---------------------------------------------------------------------------
__global__ __launch_bounds__(32)
void oproj_kernel(const _Float16* __restrict__ ao,
                  const _Float16* __restrict__ wot,
                  float* __restrict__ out) {
  const int lane = threadIdx.x, l16 = lane & 15, lg = lane >> 4;
  const int dt  = (blockIdx.x % 18) * 8;            // 8 N-tiles of 16
  const int bt0 = (blockIdx.x / 18) * 16;
  const int K = NQ_ * H_;                           // 2048

  v8f zero = {};
  v8f acc[8];
#pragma unroll
  for (int i = 0; i < 8; i++) acc[i] = zero;

  const _Float16* abase = ao + (size_t)(bt0 + l16) * K;
  for (int kk0 = 0; kk0 < K; kk0 += 32) {
    FragA fa;
#pragma unroll
    for (int r = 0; r < 8; r++)
      fa.u[r] = *(const uint32_t*)(abase + kk0 + a_kbase(r, lg));
    const _Float16* wbase = wot + (size_t)(dt * 16 + l16) * K + kk0 + 16 * lg;
    __builtin_prefetch(wbase + 32, 0, 1);
#pragma unroll
    for (int c = 0; c < 8; c++) {
      FragB fb;
      load_fragB(fb, wbase + (size_t)c * 16 * K);
      acc[c] = wmma16(fa.h, fb.h, acc[c]);
    }
  }
#pragma unroll
  for (int c = 0; c < 8; c++) {
    float* op = out + (size_t)(bt0 + 8 * lg) * D_ + (dt + c) * 16 + l16;
#pragma unroll
    for (int r = 0; r < 8; r++) op[(size_t)r * D_] = acc[c][r];
  }
}

// ---------------------------------------------------------------------------
extern "C" void kernel_launch(void* const* d_in, const int* in_sizes, int n_in,
                              void* d_out, int out_size, void* d_ws,
                              size_t ws_size, hipStream_t stream) {
  const float* x  = (const float*)d_in[0];
  // d_in[1] = attention_mask (causal) -- reimplemented analytically
  const float* Wq = (const float*)d_in[2];
  const float* Wk = (const float*)d_in[3];
  const float* Wv = (const float*)d_in[4];
  const float* Wo = (const float*)d_in[5];
  float* out = (float*)d_out;

  // workspace carve-up (f16):
  // q 16MB | k 8MB | v^T 8MB | attn-out 16MB | Wqkv^T 18.9MB | Wo^T 9.4MB
  // | x-f16 18.9MB   (total ~97MB)
  _Float16* q_h = (_Float16*)d_ws;
  const size_t qe = (size_t)B_ * T_ * NQ_ * H_;
  const size_t ke = (size_t)B_ * T_ * NKV_ * H_;
  const size_t xe = (size_t)B_ * T_ * D_;
  _Float16* k_h = q_h + qe;
  _Float16* v_t = k_h + ke;
  _Float16* ao  = v_t + ke;
  _Float16* wt  = ao + qe;                       // [16][H][D]
  _Float16* wot = wt + (size_t)16 * H_ * D_;     // [D][2048]
  _Float16* xh  = wot + (size_t)D_ * NQ_ * H_;   // [B,T,D]

  // one-shot conversions
  xconv_kernel<<<(unsigned)(xe / 1024), 256, 0, stream>>>(x, xh);
  wtrans_kernel<<<dim3(H_ / 32, D_ / 32, NQ_), 256, 0, stream>>>(
      Wq, wt, D_, H_);
  wtrans_kernel<<<dim3(H_ / 32, D_ / 32, NKV_), 256, 0, stream>>>(
      Wk, wt + (size_t)NQ_ * H_ * D_, D_, H_);
  wtrans_kernel<<<dim3(H_ / 32, D_ / 32, NKV_), 256, 0, stream>>>(
      Wv, wt + (size_t)(NQ_ + NKV_) * H_ * D_, D_, H_);
  // Wo: [NQ*H=2048][D] -> [D][2048]
  wtrans_kernel<<<dim3(D_ / 32, (NQ_ * H_) / 32, 1), 256, 0, stream>>>(
      Wo, wot, NQ_ * H_, D_);

  qkv_rope_kernel<<<B_ * 16 * (T_ / 16), 32, 0, stream>>>(xh, wt, q_h, k_h,
                                                          v_t);
  attn_kernel<<<B_ * NQ_ * (T_ / 16), 32, 0, stream>>>(q_h, k_h, v_t, ao);
  oproj_kernel<<<(B_ * T_ / 16) * (D_ / 128), 32, 0, stream>>>(ao, wot, out);
}
